// MoeBlock_27367531610983
// MI455X (gfx1250) — compile-verified, matrix operand order checked by reading, not run on hardware
//
#include <hip/hip_runtime.h>
#include <hip/hip_bf16.h>
#include <stdint.h>

// ---------------- Problem constants (match reference) ----------------
#define NEXP 8
#define HDIM 1024
#define IDIM 1024
#define NTOK 4096          // 2 * 2048
#define TOPK 2
#define KSTEP 64
#define LDS_STRIDE 72      // 64 + 8 pad halves; 144 B row stride (16B aligned)

// ---------------- Types ----------------
typedef __attribute__((ext_vector_type(16))) __bf16 bf16x16;
typedef __attribute__((ext_vector_type(8)))  float  v8f;

struct alignas(16) F4 { float x, y, z, w; };
struct alignas(16) H8 { unsigned short h[8]; };
struct alignas(8)  H4 { unsigned short h[4]; };

__device__ __forceinline__ unsigned short f2bf_bits(float f) {
    unsigned u = __builtin_bit_cast(unsigned, f);
    return (unsigned short)((u + 0x7FFFu + ((u >> 16) & 1u)) >> 16);  // RNE
}
__device__ __forceinline__ __bf16 f2bf(float f) {
    unsigned short h = f2bf_bits(f);
    return __builtin_bit_cast(__bf16, h);
}

// Load one WMMA bf16 fragment from an LDS row segment starting at k-offset `ks`.
// Lane's 16 halves are K = {ks+8*hi .. +7, ks+16+8*hi .. +7}: two 16B chunks.
__device__ __forceinline__ bf16x16 load_frag(const __bf16* row, int hi) {
    struct alignas(16) P { F4 a; F4 b; };
    P p;
    p.a = *reinterpret_cast<const F4*>(row + 8 * hi);
    p.b = *reinterpret_cast<const F4*>(row + 16 + 8 * hi);
    return __builtin_bit_cast(bf16x16, p);
}

__device__ __forceinline__ v8f wmma_bf16(bf16x16 a, bf16x16 b, v8f c) {
    return __builtin_amdgcn_wmma_f32_16x16x32_bf16(
        false, a, false, b, (short)0, c, false, false);
}

// Pack 4 k-consecutive bf16 (same n) into one 8-byte LDS store, [n][k] layout.
__device__ __forceinline__ void transpose_store4(
    __bf16 (*B)[LDS_STRIDE], const H8& r0, const H8& r1, const H8& r2, const H8& r3,
    int nn, int k4)
{
#pragma unroll
    for (int j = 0; j < 8; ++j) {
        unsigned long long v =
            (unsigned long long)r0.h[j]
          | ((unsigned long long)r1.h[j] << 16)
          | ((unsigned long long)r2.h[j] << 32)
          | ((unsigned long long)r3.h[j] << 48);
        *reinterpret_cast<unsigned long long*>(&B[nn + j][k4]) = v;
    }
}

// ---------------- Kernel 0: zero expert counters ----------------
__global__ void moe_zero_kernel(int* __restrict__ cnt) {
    if (threadIdx.x < NEXP) cnt[threadIdx.x] = 0;
}

// ---------------- Kernel 0b: bulk f32 -> bf16 conversion (4 elems/thread) ---
__global__ __launch_bounds__(256) void moe_cvt_kernel(
    const float* __restrict__ src, __bf16* __restrict__ dst, int n4)
{
    int i = blockIdx.x * 256 + threadIdx.x;
    if (i >= n4) return;
    const F4 v = reinterpret_cast<const F4*>(src)[i];
    H4 o;
    o.h[0] = f2bf_bits(v.x); o.h[1] = f2bf_bits(v.y);
    o.h[2] = f2bf_bits(v.z); o.h[3] = f2bf_bits(v.w);
    reinterpret_cast<H4*>(dst)[i] = o;
}

// ---------------- Kernel 1: router (softmax + top-2, build expert lists) ----
__global__ __launch_bounds__(256) void moe_router_kernel(
    const float* __restrict__ x, const float* __restrict__ gw,
    const float* __restrict__ gb,
    int* __restrict__ cnt, int* __restrict__ tok, int* __restrict__ gslot,
    float* __restrict__ w2)
{
    const int t = blockIdx.x * 8 + (threadIdx.x >> 5);
    const int lane = threadIdx.x & 31;
    if (t >= NTOK) return;

    float acc[NEXP];
#pragma unroll
    for (int e = 0; e < NEXP; ++e) acc[e] = 0.0f;

    const float* xr = x + (size_t)t * HDIM;
    for (int h = lane; h < HDIM; h += 32) {
        float xv = xr[h];
        const F4 g0 = *reinterpret_cast<const F4*>(gw + (size_t)h * NEXP);
        const F4 g1 = *reinterpret_cast<const F4*>(gw + (size_t)h * NEXP + 4);
        acc[0] += xv * g0.x; acc[1] += xv * g0.y;
        acc[2] += xv * g0.z; acc[3] += xv * g0.w;
        acc[4] += xv * g1.x; acc[5] += xv * g1.y;
        acc[6] += xv * g1.z; acc[7] += xv * g1.w;
    }
#pragma unroll
    for (int e = 0; e < NEXP; ++e)
#pragma unroll
        for (int m = 16; m >= 1; m >>= 1)
            acc[e] += __shfl_xor(acc[e], m, 32);

    if (lane == 0) {
        float p[NEXP];
        float mx = -1e30f;
#pragma unroll
        for (int e = 0; e < NEXP; ++e) { p[e] = acc[e] + gb[e]; mx = fmaxf(mx, p[e]); }
        float s = 0.0f;
#pragma unroll
        for (int e = 0; e < NEXP; ++e) { p[e] = __expf(p[e] - mx); s += p[e]; }
        float inv = 1.0f / s;
#pragma unroll
        for (int e = 0; e < NEXP; ++e) p[e] *= inv;

        int e0 = 0;
#pragma unroll
        for (int e = 1; e < NEXP; ++e) if (p[e] > p[e0]) e0 = e;   // ties -> lowest idx
        int e1 = (e0 == 0) ? 1 : 0;
#pragma unroll
        for (int e = 0; e < NEXP; ++e) if (e != e0 && p[e] > p[e1]) e1 = e;

        int s0 = atomicAdd(&cnt[e0], 1);
        tok[e0 * NTOK + s0] = t;  gslot[e0 * NTOK + s0] = 2 * t;
        int s1 = atomicAdd(&cnt[e1], 1);
        tok[e1 * NTOK + s1] = t;  gslot[e1 * NTOK + s1] = 2 * t + 1;
        w2[2 * t]     = p[e0];
        w2[2 * t + 1] = p[e1];
    }
}

// ---------------- Kernel 2: up projection (gathered, fused gate+up+SiLU) ----
// block = 256 thr = 8 waves (4x2), block tile 128(M) x 128(N), K step 64.
__global__ __launch_bounds__(256) void moe_up_kernel(
    const __bf16* __restrict__ xbf,
    const __bf16* __restrict__ Wg, const float* __restrict__ bgv,
    const __bf16* __restrict__ Wu, const float* __restrict__ buv,
    const int* __restrict__ cnt, const int* __restrict__ tok,
    const int* __restrict__ gslot, __bf16* __restrict__ inter)
{
    const int e = blockIdx.z;
    const int cntE = cnt[e];
    const int m0 = blockIdx.y * 128;
    if (m0 >= cntE) return;
    const int n0 = blockIdx.x * 128;

    __shared__ alignas(16) __bf16 As[128][LDS_STRIDE];
    __shared__ alignas(16) __bf16 Bg[128][LDS_STRIDE];   // [n][k]
    __shared__ alignas(16) __bf16 Bu[128][LDS_STRIDE];

    const int tid  = threadIdx.x;
    const int lane = tid & 31;
    const int wave = tid >> 5;
    const int wm = wave & 3, wn = wave >> 2;     // 4 x 2 waves
    const int hi = lane >> 4, ln = lane & 15;

    // tokens for the 4 A-load passes (row = tid/8 + 32*p, 8 halves each)
    int tokRow[4];
#pragma unroll
    for (int p = 0; p < 4; ++p) {
        int idx = m0 + (tid >> 3) + 32 * p;
        tokRow[p] = (idx < cntE) ? tok[e * NTOK + idx] : tok[e * NTOK];
    }
    const int kg = tid >> 4;            // 0..15 -> k4 = kg*4
    const int nn = (tid & 15) * 8;      // 0..120

    v8f accG[2][4] = {};
    v8f accU[2][4] = {};

    const __bf16* Wge = Wg + (size_t)e * HDIM * IDIM;
    const __bf16* Wue = Wu + (size_t)e * HDIM * IDIM;

    for (int k0 = 0; k0 < HDIM; k0 += KSTEP) {
        // --- A tile: gathered bf16 x rows, straight 16B copies ---
#pragma unroll
        for (int p = 0; p < 4; ++p) {
            int s = (tid >> 3) + 32 * p;
            const F4 v = *reinterpret_cast<const F4*>(
                xbf + (size_t)tokRow[p] * HDIM + k0 + (tid & 7) * 8);
            *reinterpret_cast<F4*>(&As[s][(tid & 7) * 8]) = v;
        }
        // --- B tiles: 4 k-rows x 8 n each, register repack -> b64 stores ---
        {
            const int k4 = kg * 4;
            const size_t base = (size_t)(k0 + k4) * IDIM + n0 + nn;
            const H8 g0 = *reinterpret_cast<const H8*>(Wge + base);
            const H8 g1 = *reinterpret_cast<const H8*>(Wge + base + IDIM);
            const H8 g2 = *reinterpret_cast<const H8*>(Wge + base + 2 * IDIM);
            const H8 g3 = *reinterpret_cast<const H8*>(Wge + base + 3 * IDIM);
            transpose_store4(Bg, g0, g1, g2, g3, nn, k4);
            const H8 u0 = *reinterpret_cast<const H8*>(Wue + base);
            const H8 u1 = *reinterpret_cast<const H8*>(Wue + base + IDIM);
            const H8 u2 = *reinterpret_cast<const H8*>(Wue + base + 2 * IDIM);
            const H8 u3 = *reinterpret_cast<const H8*>(Wue + base + 3 * IDIM);
            transpose_store4(Bu, u0, u1, u2, u3, nn, k4);
        }
        __syncthreads();

#pragma unroll
        for (int ks = 0; ks < KSTEP; ks += 32) {
            bf16x16 aF[2];
#pragma unroll
            for (int mt = 0; mt < 2; ++mt)
                aF[mt] = load_frag(&As[wm * 32 + mt * 16 + ln][ks], hi);
#pragma unroll
            for (int nt = 0; nt < 4; ++nt) {
                bf16x16 bG = load_frag(&Bg[wn * 64 + nt * 16 + ln][ks], hi);
                bf16x16 bU = load_frag(&Bu[wn * 64 + nt * 16 + ln][ks], hi);
#pragma unroll
                for (int mt = 0; mt < 2; ++mt) {
                    accG[mt][nt] = wmma_bf16(aF[mt], bG, accG[mt][nt]);
                    accU[mt][nt] = wmma_bf16(aF[mt], bU, accU[mt][nt]);
                }
            }
        }
        __syncthreads();
    }

    // --- epilogue: bias + SiLU(g)*u -> bf16 inter[2t+k] ---
    const float* bgp = bgv + e * IDIM;
    const float* bup = buv + e * IDIM;
#pragma unroll
    for (int mt = 0; mt < 2; ++mt) {
#pragma unroll
        for (int v = 0; v < 8; ++v) {
            int idx = m0 + wm * 32 + mt * 16 + 8 * hi + v;   // D row = v + 8*hi
            if (idx < cntE) {
                int gs = gslot[e * NTOK + idx];
                __bf16* orow = inter + (size_t)gs * IDIM;
#pragma unroll
                for (int nt = 0; nt < 4; ++nt) {
                    int col = n0 + wn * 64 + nt * 16 + ln;
                    float g = accG[mt][nt][v] + bgp[col];
                    float u = accU[mt][nt][v] + bup[col];
                    float si = g * (1.0f / (1.0f + __expf(-g)));
                    orow[col] = f2bf(si * u);
                }
            }
        }
    }
}

// ---------------- Kernel 3: down projection ----------------
__global__ __launch_bounds__(256) void moe_down_kernel(
    const __bf16* __restrict__ inter,
    const __bf16* __restrict__ Wd, const float* __restrict__ bdv,
    const int* __restrict__ cnt, const int* __restrict__ gslot,
    float* __restrict__ yslot)
{
    const int e = blockIdx.z;
    const int cntE = cnt[e];
    const int m0 = blockIdx.y * 128;
    if (m0 >= cntE) return;
    const int n0 = blockIdx.x * 128;

    __shared__ alignas(16) __bf16 As[128][LDS_STRIDE];
    __shared__ alignas(16) __bf16 Bd[128][LDS_STRIDE];

    const int tid  = threadIdx.x;
    const int lane = tid & 31;
    const int wave = tid >> 5;
    const int wm = wave & 3, wn = wave >> 2;
    const int hi = lane >> 4, ln = lane & 15;

    int gsRow[4];
#pragma unroll
    for (int p = 0; p < 4; ++p) {
        int idx = m0 + (tid >> 3) + 32 * p;
        gsRow[p] = (idx < cntE) ? gslot[e * NTOK + idx] : gslot[e * NTOK];
    }
    const int kg = tid >> 4;
    const int nn = (tid & 15) * 8;

    v8f acc[2][4] = {};
    const __bf16* Wde = Wd + (size_t)e * IDIM * HDIM;

    for (int k0 = 0; k0 < IDIM; k0 += KSTEP) {
        // --- A tile: gathered bf16 inter rows ---
#pragma unroll
        for (int p = 0; p < 4; ++p) {
            int s = (tid >> 3) + 32 * p;
            const F4 v = *reinterpret_cast<const F4*>(
                inter + (size_t)gsRow[p] * IDIM + k0 + (tid & 7) * 8);
            *reinterpret_cast<F4*>(&As[s][(tid & 7) * 8]) = v;
        }
        // --- B tile: Wd bf16, register repack -> b64 stores ---
        {
            const int k4 = kg * 4;
            const size_t base = (size_t)(k0 + k4) * HDIM + n0 + nn;
            const H8 d0 = *reinterpret_cast<const H8*>(Wde + base);
            const H8 d1 = *reinterpret_cast<const H8*>(Wde + base + HDIM);
            const H8 d2 = *reinterpret_cast<const H8*>(Wde + base + 2 * HDIM);
            const H8 d3 = *reinterpret_cast<const H8*>(Wde + base + 3 * HDIM);
            transpose_store4(Bd, d0, d1, d2, d3, nn, k4);
        }
        __syncthreads();

#pragma unroll
        for (int ks = 0; ks < KSTEP; ks += 32) {
            bf16x16 aF[2];
#pragma unroll
            for (int mt = 0; mt < 2; ++mt)
                aF[mt] = load_frag(&As[wm * 32 + mt * 16 + ln][ks], hi);
#pragma unroll
            for (int nt = 0; nt < 4; ++nt) {
                bf16x16 bF = load_frag(&Bd[wn * 64 + nt * 16 + ln][ks], hi);
#pragma unroll
                for (int mt = 0; mt < 2; ++mt)
                    acc[mt][nt] = wmma_bf16(aF[mt], bF, acc[mt][nt]);
            }
        }
        __syncthreads();
    }

    const float* bdp = bdv + e * HDIM;
#pragma unroll
    for (int mt = 0; mt < 2; ++mt) {
#pragma unroll
        for (int v = 0; v < 8; ++v) {
            int idx = m0 + wm * 32 + mt * 16 + 8 * hi + v;
            if (idx < cntE) {
                int gs = gslot[e * NTOK + idx];
                float* orow = yslot + (size_t)gs * HDIM;
#pragma unroll
                for (int nt = 0; nt < 4; ++nt) {
                    int col = n0 + wn * 64 + nt * 16 + ln;
                    orow[col] = acc[mt][nt][v] + bdp[col];
                }
            }
        }
    }
}

// ---------------- Kernel 4: weighted combine (no atomics) ----------------
__global__ __launch_bounds__(256) void moe_combine_kernel(
    const float* __restrict__ yslot, const float* __restrict__ w2,
    float* __restrict__ out)
{
    const size_t i = (size_t)blockIdx.x * 256 + threadIdx.x;  // over T*H/4
    const int t  = (int)(i >> 8);          // H/4 = 256 float4 per token
    const int c4 = (int)(i & 255);
    const float w0 = w2[2 * t];
    const float w1 = w2[2 * t + 1];
    const F4 a = reinterpret_cast<const F4*>(yslot)[(size_t)(2 * t) * 256 + c4];
    const F4 b = reinterpret_cast<const F4*>(yslot)[(size_t)(2 * t + 1) * 256 + c4];
    F4 r;
    r.x = w0 * a.x + w1 * b.x;
    r.y = w0 * a.y + w1 * b.y;
    r.z = w0 * a.z + w1 * b.z;
    r.w = w0 * a.w + w1 * b.w;
    reinterpret_cast<F4*>(out)[i] = r;
}

// ---------------- Host launcher ----------------
extern "C" void kernel_launch(void* const* d_in, const int* in_sizes, int n_in,
                              void* d_out, int out_size, void* d_ws, size_t ws_size,
                              hipStream_t stream) {
    const float* x   = (const float*)d_in[0];   // [2,2048,1024]
    const float* gw  = (const float*)d_in[1];   // [1024,8]
    const float* gb  = (const float*)d_in[2];   // [8]
    const float* Wg  = (const float*)d_in[3];   // [8,1024,1024]
    const float* bg  = (const float*)d_in[4];   // [8,1024]
    const float* Wu  = (const float*)d_in[5];
    const float* bu  = (const float*)d_in[6];
    const float* Wd  = (const float*)d_in[7];   // [8,1024,1024]
    const float* bd  = (const float*)d_in[8];
    (void)in_sizes; (void)n_in; (void)out_size; (void)ws_size;

    // Workspace layout
    const size_t W_ELEMS = (size_t)NEXP * HDIM * IDIM;    // 8M
    const size_t X_ELEMS = (size_t)NTOK * HDIM;           // 4M
    char* ws = (char*)d_ws;
    size_t off = 0;
    int*    cnt   = (int*)(ws + off);  off += 256;
    int*    tok   = (int*)(ws + off);  off += (size_t)NEXP * NTOK * 4;
    int*    gsl   = (int*)(ws + off);  off += (size_t)NEXP * NTOK * 4;
    float*  w2    = (float*)(ws + off); off += (size_t)2 * NTOK * 4;
    __bf16* inter = (__bf16*)(ws + off); off += (size_t)2 * NTOK * IDIM * 2;  // 16 MiB
    float*  yslot = (float*)(ws + off);  off += (size_t)2 * NTOK * HDIM * 4;  // 32 MiB
    __bf16* xbf   = (__bf16*)(ws + off); off += X_ELEMS * 2;                  //  8 MiB
    __bf16* Wgbf  = (__bf16*)(ws + off); off += W_ELEMS * 2;                  // 16 MiB
    __bf16* Wubf  = (__bf16*)(ws + off); off += W_ELEMS * 2;                  // 16 MiB
    __bf16* Wdbf  = (__bf16*)(ws + off); off += W_ELEMS * 2;                  // 16 MiB

    moe_zero_kernel<<<1, 32, 0, stream>>>(cnt);

    // One-time f32 -> bf16 mirrors (memory bound, ~170 MB total traffic)
    {
        int n4w = (int)(W_ELEMS / 4), n4x = (int)(X_ELEMS / 4);
        moe_cvt_kernel<<<(n4x + 255) / 256, 256, 0, stream>>>(x,  xbf,  n4x);
        moe_cvt_kernel<<<(n4w + 255) / 256, 256, 0, stream>>>(Wg, Wgbf, n4w);
        moe_cvt_kernel<<<(n4w + 255) / 256, 256, 0, stream>>>(Wu, Wubf, n4w);
        moe_cvt_kernel<<<(n4w + 255) / 256, 256, 0, stream>>>(Wd, Wdbf, n4w);
    }

    moe_router_kernel<<<NTOK / 8, 256, 0, stream>>>(x, gw, gb, cnt, tok, gsl, w2);

    {
        dim3 grid(IDIM / 128, NTOK / 128, NEXP);   // (8, 32, 8)
        moe_up_kernel<<<grid, 256, 0, stream>>>(xbf, Wgbf, bg, Wubf, bu,
                                                cnt, tok, gsl, inter);
    }
    {
        dim3 grid(HDIM / 128, NTOK / 128, NEXP);   // (8, 32, 8)
        moe_down_kernel<<<grid, 256, 0, stream>>>(inter, Wdbf, bd, cnt, gsl, yslot);
    }
    moe_combine_kernel<<<(NTOK * HDIM / 4) / 256, 256, 0, stream>>>(yslot, w2, (float*)d_out);
}
